// ANsDiscovery_32856499814405
// MI455X (gfx1250) — matrix operand order repaired and checked by reading, not compiled, not run on hardware
//
#include <hip/hip_runtime.h>
#include <stdint.h>

#define N_SAMPLES 20000
#define DIM       128
#define ANS       5000
#define APAD      5008            // anchors padded to multiple of 16
#define ANS_SIZE  10
#define INV_T     (1.0f/0.07f)    // max possible logit (normalized features)
#define NCOLTILES (N_SAMPLES/16)  // 1250
#define SORT_SZ   32768

typedef __bf16 bf16_t;
typedef __attribute__((ext_vector_type(16))) __bf16 v16bf;
typedef __attribute__((ext_vector_type(8)))  __bf16 v8bf;
typedef __attribute__((ext_vector_type(8)))  float  v8f;

// ---------------- fragment loaders (CDNA5 WMMA 16x16x32 bf16 layouts) --------
// A (16x32, MxK): lane l holds row M=l&15; lanes<16 -> K {k0..k0+7, k0+16..k0+23},
// lanes>=16 -> +8. Two contiguous 16B runs per lane.
__device__ __forceinline__ v16bf load_a_frag(const uint16_t* feats, int row,
                                             int k0, int laneHi) {
  const bf16_t* base =
      reinterpret_cast<const bf16_t*>(feats + row * DIM + k0 + (laneHi ? 8 : 0));
  v8bf lo = *reinterpret_cast<const v8bf*>(base);
  v8bf hi = *reinterpret_cast<const v8bf*>(base + 16);
  return __builtin_shufflevector(lo, hi, 0,1,2,3,4,5,6,7,8,9,10,11,12,13,14,15);
}
// B (32x16, KxN): lane l holds column N=l&15; lanes<16 -> K {k0..k0+7, k0+16..k0+23},
// lanes>=16 -> +8. Two contiguous 16B runs per lane.
__device__ __forceinline__ v16bf load_b_frag(const uint16_t* feats, int col,
                                             int k0, int laneHi) {
  const bf16_t* base =
      reinterpret_cast<const bf16_t*>(feats + col * DIM + k0 + (laneHi ? 8 : 0));
  v8bf lo = *reinterpret_cast<const v8bf*>(base);
  v8bf hi = *reinterpret_cast<const v8bf*>(base + 16);
  return __builtin_shufflevector(lo, hi, 0,1,2,3,4,5,6,7,8,9,10,11,12,13,14,15);
}

// ---------------- K0: f32 -> bf16 (RNE) --------------------------------------
__global__ void convert_bf16_kernel(const float* __restrict__ in,
                                    uint16_t* __restrict__ out, int n) {
  int i = blockIdx.x * blockDim.x + threadIdx.x;
  if (i < n) {
    uint32_t x = __float_as_uint(in[i]);
    uint32_t r = (x + 0x7FFFu + ((x >> 16) & 1u)) >> 16;
    out[i] = (uint16_t)r;
  }
}

// ---------------- K1: streaming WMMA entropy ---------------------------------
// 8 waves/block, each wave owns a 16-row tile (128 rows/block) and walks all
// column tiles; B tiles are shared across waves via WGP$ (same addresses).
// H_i = log(S_i) + M - T_i/S_i with S=sum exp(z-M), T=sum exp(z-M)*z, M=1/TEMP.
__global__ void __launch_bounds__(256)
entropy_kernel(const uint16_t* __restrict__ feats, float* __restrict__ entropy_out) {
  const int tid = threadIdx.x;
  const int wave = tid >> 5, lane = tid & 31;
  const int laneHi = lane >> 4, laneLo = lane & 15;
  const int rowBase = blockIdx.x * 128 + wave * 16;

  int arow = rowBase + laneLo;
  if (arow >= N_SAMPLES) arow = N_SAMPLES - 1;  // clamped dup rows, writes guarded
  v16bf a_frag[4];
#pragma unroll
  for (int ks = 0; ks < 4; ks++) a_frag[ks] = load_a_frag(feats, arow, ks * 32, laneHi);

  float S[8], T[8];
#pragma unroll
  for (int r = 0; r < 8; r++) { S[r] = 0.f; T[r] = 0.f; }

  // double-buffered B fragments: next tile's loads overlap current WMMA chain
  int col0 = laneLo;
  v16bf b0 = load_b_frag(feats, col0, 0, laneHi);
  v16bf b1 = load_b_frag(feats, col0, 32, laneHi);
  v16bf b2 = load_b_frag(feats, col0, 64, laneHi);
  v16bf b3 = load_b_frag(feats, col0, 96, laneHi);

  for (int jt = 0; jt < NCOLTILES; jt++) {
    int njt = jt + 1 < NCOLTILES ? jt + 1 : jt;      // clamped -> branch-free
    int ncol = njt * 16 + laneLo;
    int pjt = jt + 8 < NCOLTILES ? jt + 8 : jt;
    __builtin_prefetch(feats + (pjt * 16 + laneLo) * DIM, 0, 1);
    v16bf n0 = load_b_frag(feats, ncol, 0, laneHi);
    v16bf n1 = load_b_frag(feats, ncol, 32, laneHi);
    v16bf n2 = load_b_frag(feats, ncol, 64, laneHi);
    v16bf n3 = load_b_frag(feats, ncol, 96, laneHi);

    v8f c = {};
    c = __builtin_amdgcn_wmma_f32_16x16x32_bf16(false, a_frag[0], false, b0, (short)0, c, false, false);
    c = __builtin_amdgcn_wmma_f32_16x16x32_bf16(false, a_frag[1], false, b1, (short)0, c, false, false);
    c = __builtin_amdgcn_wmma_f32_16x16x32_bf16(false, a_frag[2], false, b2, (short)0, c, false, false);
    c = __builtin_amdgcn_wmma_f32_16x16x32_bf16(false, a_frag[3], false, b3, (short)0, c, false, false);

#pragma unroll
    for (int r = 0; r < 8; r++) {
      float z = c[r] * INV_T;
      float e = __expf(z - INV_T);
      S[r] += e;
      T[r] += e * z;
    }
    b0 = n0; b1 = n1; b2 = n2; b3 = n3;
  }

  // reduce over the 16 lanes holding the same rows (xor shuffles, width 16)
#pragma unroll
  for (int r = 0; r < 8; r++) {
#pragma unroll
    for (int m = 1; m < 16; m <<= 1) {
      S[r] += __shfl_xor(S[r], m, 16);
      T[r] += __shfl_xor(T[r], m, 16);
    }
  }
  if (laneLo == 0) {
#pragma unroll
    for (int r = 0; r < 8; r++) {
      int row = rowBase + r + 8 * laneHi;
      if (row < N_SAMPLES)
        entropy_out[row] = __logf(S[r]) + INV_T - T[r] / S[r];
    }
  }
}

// ---------------- K2: single-block bitonic sort in 256KB LDS -----------------
// key packed as (entropy_bits<<32)|index: entropy>=0 so IEEE bits are order-
// preserving, and low index bits give the ascending-index tie-break for free.
__global__ void __launch_bounds__(1024)
sort_kernel(const float* __restrict__ entropy, int* __restrict__ anchor_idx,
            int* __restrict__ mask, int* __restrict__ anchor_pos,
            float* __restrict__ out_anchor) {
  extern __shared__ unsigned char smem[];
  unsigned long long* arr = reinterpret_cast<unsigned long long*>(smem);
  const int tid = threadIdx.x;

  for (int i = tid; i < SORT_SZ; i += 1024) {
    if (i < N_SAMPLES)
      arr[i] = ((unsigned long long)__float_as_uint(entropy[i]) << 32) | (uint32_t)i;
    else
      arr[i] = ~0ULL;
  }
  for (int i = tid; i < N_SAMPLES; i += 1024) mask[i] = 1;
  __syncthreads();

  for (int k = 2; k <= SORT_SZ; k <<= 1) {
    for (int j = k >> 1; j > 0; j >>= 1) {
      for (int i = tid; i < SORT_SZ; i += 1024) {
        int ixj = i ^ j;
        if (ixj > i) {
          bool up = ((i & k) == 0);
          unsigned long long a = arr[i], b = arr[ixj];
          if ((a > b) == up) { arr[i] = b; arr[ixj] = a; }
        }
      }
      __syncthreads();
    }
  }
  for (int r = tid; r < ANS; r += 1024) {
    int a = (int)(arr[r] & 0xffffffffu);
    anchor_idx[r] = a;
    out_anchor[r] = (float)a;
    mask[a] = 0;
    anchor_pos[a] = r;
  }
}

// ---------------- K3: prefix scan -> instance list + positions ---------------
__global__ void __launch_bounds__(1024)
scan_kernel(const int* __restrict__ mask, const int* __restrict__ anchor_pos,
            float* __restrict__ out_inst, float* __restrict__ out_pos) {
  __shared__ int part[1024];
  const int tid = threadIdx.x;
  const int CH = 20;  // 1024*20 >= 20000
  const int start = tid * CH;
  int s = 0;
  for (int i = 0; i < CH; i++) { int g = start + i; if (g < N_SAMPLES) s += mask[g]; }
  part[tid] = s;
  __syncthreads();
  for (int off = 1; off < 1024; off <<= 1) {
    int v = (tid >= off) ? part[tid - off] : 0;
    __syncthreads();
    part[tid] += v;
    __syncthreads();
  }
  int cum = (tid > 0) ? part[tid - 1] : 0;
  for (int i = 0; i < CH; i++) {
    int g = start + i;
    if (g >= N_SAMPLES) break;
    int m = mask[g];
    cum += m;
    if (m) { out_inst[cum - 1] = (float)g; out_pos[g] = (float)(-cum); }
    else   { out_pos[g] = (float)anchor_pos[g]; }
  }
}

// ---------------- K4: gather anchor features (bf16, zero-padded) -------------
__global__ void gather_kernel(const uint16_t* __restrict__ feats,
                              const int* __restrict__ anchor_idx,
                              uint16_t* __restrict__ anchor_feats) {
  int i = blockIdx.x * blockDim.x + threadIdx.x;
  if (i >= APAD * DIM) return;
  int a = i / DIM, k = i % DIM;
  anchor_feats[i] = (a < ANS) ? feats[anchor_idx[a] * DIM + k] : (uint16_t)0;
}

// ---------------- K5: WMMA neighbour sims + per-wave streaming top-10 --------
// 8 waves/block, each wave owns a 16-anchor tile; per-wave private LDS tile is
// transposed after WMMA, then lanes 0-15 of each wave maintain a register top-10
// (fully unrolled shift network, no scratch spill; strict '>' keeps lowest index).
__global__ void __launch_bounds__(256)
neighbor_kernel(const uint16_t* __restrict__ feats,
                const uint16_t* __restrict__ anchor_feats,
                const int* __restrict__ anchor_idx,
                float* __restrict__ out_neigh, int* __restrict__ neigh_ws) {
  __shared__ float sims[8][16][17];
  const int tid = threadIdx.x;
  const int wave = tid >> 5, lane = tid & 31;
  const int laneHi = lane >> 4, laneLo = lane & 15;
  const int aBase = blockIdx.x * 128 + wave * 16;

  int arow = aBase + laneLo;
  if (arow >= APAD) arow = APAD - 1;
  v16bf a_frag[4];
#pragma unroll
  for (int ks = 0; ks < 4; ks++)
    a_frag[ks] = load_a_frag(anchor_feats, arow, ks * 32, laneHi);

  const int myA = aBase + laneLo;                  // row scanned by lanes 0-15
  int self = -1;
  if (lane < 16 && myA < ANS) self = anchor_idx[myA];

  float bestV[ANS_SIZE];
  int bestI[ANS_SIZE];
#pragma unroll
  for (int k = 0; k < ANS_SIZE; k++) { bestV[k] = -1.0e30f; bestI[k] = 0x7fffffff; }

  int col0 = laneLo;
  v16bf b0 = load_b_frag(feats, col0, 0, laneHi);
  v16bf b1 = load_b_frag(feats, col0, 32, laneHi);
  v16bf b2 = load_b_frag(feats, col0, 64, laneHi);
  v16bf b3 = load_b_frag(feats, col0, 96, laneHi);

  for (int jt = 0; jt < NCOLTILES; jt++) {
    int njt = jt + 1 < NCOLTILES ? jt + 1 : jt;
    int ncol = njt * 16 + laneLo;
    v16bf n0 = load_b_frag(feats, ncol, 0, laneHi);
    v16bf n1 = load_b_frag(feats, ncol, 32, laneHi);
    v16bf n2 = load_b_frag(feats, ncol, 64, laneHi);
    v16bf n3 = load_b_frag(feats, ncol, 96, laneHi);

    v8f c = {};
    c = __builtin_amdgcn_wmma_f32_16x16x32_bf16(false, a_frag[0], false, b0, (short)0, c, false, false);
    c = __builtin_amdgcn_wmma_f32_16x16x32_bf16(false, a_frag[1], false, b1, (short)0, c, false, false);
    c = __builtin_amdgcn_wmma_f32_16x16x32_bf16(false, a_frag[2], false, b2, (short)0, c, false, false);
    c = __builtin_amdgcn_wmma_f32_16x16x32_bf16(false, a_frag[3], false, b3, (short)0, c, false, false);

#pragma unroll
    for (int r = 0; r < 8; r++) sims[wave][r + 8 * laneHi][laneLo] = c[r];
    __syncthreads();  // uniform trip count across all waves

    if (lane < 16) {
      const int gBase = jt * 16;
#pragma unroll
      for (int n = 0; n < 16; n++) {      // ascending column order
        int g = gBase + n;
        float v = sims[wave][laneLo][n];
        if (g == self) v = -1.0f;
        if (v > bestV[ANS_SIZE - 1]) {    // strict > => lowest-index tie-break
#pragma unroll
          for (int k = ANS_SIZE - 1; k >= 1; k--) {
            if (v > bestV[k]) {
              bool sh = v > bestV[k - 1];
              bestV[k] = sh ? bestV[k - 1] : v;
              bestI[k] = sh ? bestI[k - 1] : g;
            }
          }
          if (v > bestV[0]) { bestV[0] = v; bestI[0] = g; }
        }
      }
    }
    __syncthreads();
    b0 = n0; b1 = n1; b2 = n2; b3 = n3;
  }

  if (lane < 16 && myA < ANS) {
#pragma unroll
    for (int k = 0; k < ANS_SIZE; k++) {
      out_neigh[myA * ANS_SIZE + k] = (float)bestI[k];
      neigh_ws[myA * ANS_SIZE + k] = bestI[k];
    }
  }
}

// ---------------- K6: consistency reduction ----------------------------------
__global__ void __launch_bounds__(1024)
consistency_kernel(const int* __restrict__ labels, const int* __restrict__ anchor_idx,
                   const int* __restrict__ neigh, float* __restrict__ out) {
  __shared__ float red[1024];
  const int tid = threadIdx.x;
  float s = 0.f;
  for (int i = tid; i < ANS * ANS_SIZE; i += 1024) {
    int a = i / ANS_SIZE;
    s += (labels[anchor_idx[a]] == labels[neigh[i]]) ? 1.f : 0.f;
  }
  red[tid] = s;
  __syncthreads();
  for (int off = 512; off > 0; off >>= 1) {
    if (tid < off) red[tid] += red[tid + off];
    __syncthreads();
  }
  if (tid == 0) out[0] = red[0] / (float)(ANS * ANS_SIZE);
}

// ---------------- host launcher ----------------------------------------------
extern "C" void kernel_launch(void* const* d_in, const int* in_sizes, int n_in,
                              void* d_out, int out_size, void* d_ws, size_t ws_size,
                              hipStream_t stream) {
  const float* features = (const float*)d_in[0];  // [20000,128] f32
  const int* labels = (const int*)d_in[1];        // [20000] i32
  (void)in_sizes; (void)n_in; (void)out_size; (void)ws_size;

  // d_out layout (floats): entropy[20000] | anchors[5000] | instances[15000] |
  //                        position[20000] | neighbours[50000] | consistency[1]
  float* out = (float*)d_out;
  float* out_entropy = out;
  float* out_anchor = out + 20000;
  float* out_inst = out + 25000;
  float* out_pos = out + 40000;
  float* out_neigh = out + 60000;
  float* out_cons = out + 110000;

  uint8_t* ws = (uint8_t*)d_ws;
  size_t off = 0;
  auto carve = [&](size_t bytes) {
    uint8_t* p = ws + off;
    off = (off + bytes + 255) & ~(size_t)255;
    return p;
  };
  uint16_t* feats_bf16 = (uint16_t*)carve((size_t)N_SAMPLES * DIM * 2);
  uint16_t* anchor_feats = (uint16_t*)carve((size_t)APAD * DIM * 2);
  int* anchor_idx = (int*)carve((size_t)ANS * 4);
  int* mask = (int*)carve((size_t)N_SAMPLES * 4);
  int* anchor_pos = (int*)carve((size_t)N_SAMPLES * 4);
  int* neigh_ws = (int*)carve((size_t)ANS * ANS_SIZE * 4);

  const int nconv = N_SAMPLES * DIM;
  convert_bf16_kernel<<<(nconv + 255) / 256, 256, 0, stream>>>(features, feats_bf16,
                                                               nconv);
  entropy_kernel<<<(N_SAMPLES + 127) / 128, 256, 0, stream>>>(feats_bf16, out_entropy);
  sort_kernel<<<1, 1024, SORT_SZ * 8, stream>>>(out_entropy, anchor_idx, mask,
                                                anchor_pos, out_anchor);
  scan_kernel<<<1, 1024, 0, stream>>>(mask, anchor_pos, out_inst, out_pos);
  const int ngather = APAD * DIM;
  gather_kernel<<<(ngather + 255) / 256, 256, 0, stream>>>(feats_bf16, anchor_idx,
                                                           anchor_feats);
  neighbor_kernel<<<(APAD + 127) / 128, 256, 0, stream>>>(feats_bf16, anchor_feats,
                                                          anchor_idx, out_neigh,
                                                          neigh_ws);
  consistency_kernel<<<1, 1024, 0, stream>>>(labels, anchor_idx, neigh_ws, out_cons);
}
